// EfficientUltraBlock_26542897889393
// MI455X (gfx1250) — matrix operand order repaired and checked by reading, compiled, not run on hardware
//
#include <hip/hip_runtime.h>
#include <cstdint>
#include <cstddef>

// Problem dims (fixed by setup_inputs)
#define B_  16
#define T_  512
#define D_  1024
#define H_  4
#define HD_ 256
#define R_  16
#define M_  (B_ * T_)   // 8192

typedef __attribute__((ext_vector_type(16))) __bf16 v16bf;
typedef __attribute__((ext_vector_type(8)))  float  v8f;
typedef __attribute__((ext_vector_type(4))) unsigned int u32x4;
typedef __attribute__((ext_vector_type(8))) int          i32x8;
typedef __attribute__((ext_vector_type(4))) int          i32x4;

#if defined(__HIP_DEVICE_COMPILE__) && __has_builtin(__builtin_amdgcn_tensor_load_to_lds)
#define USE_TDM 1
#else
#define USE_TDM 0
#endif

// ---------------- helpers ----------------
__device__ __forceinline__ unsigned short f2bf(float f) {
    unsigned u = __float_as_uint(f);
    if ((u & 0x7F800000u) == 0x7F800000u) {          // inf / nan
        unsigned short h = (unsigned short)(u >> 16);
        if ((u & 0x007FFFFFu) && !(h & 0x7F)) h |= 1; // keep NaN a NaN
        return h;
    }
    u += 0x7FFFu + ((u >> 16) & 1u);                  // round-to-nearest-even
    return (unsigned short)(u >> 16);
}

__device__ __forceinline__ float nan2num(float v) {
    unsigned u = __float_as_uint(v);
    if ((u & 0x7F800000u) == 0x7F800000u) {
        if (u & 0x007FFFFFu) return 0.0f;                               // NaN -> 0
        return (u & 0x80000000u) ? -3.4028234663852886e38f
                                 :  3.4028234663852886e38f;             // +-inf -> +-max
    }
    return v;
}

// ---------------- elementwise prep: xt = tanh(0.5*x), f32 + bf16 copies ----------------
__global__ __launch_bounds__(256)
void prep_xt_kernel(const float* __restrict__ x, float* __restrict__ xt_f,
                    unsigned short* __restrict__ xt_b) {
    size_t base = (size_t)blockIdx.x * 1024 + threadIdx.x;
    #pragma unroll
    for (int k = 0; k < 4; ++k) {
        size_t i = base + (size_t)k * 256;
        float v = tanhf(x[i] * 0.5f);
        xt_f[i] = v;
        xt_b[i] = f2bf(v);
    }
}

// ---------------- f32 -> bf16 cast (weights) ----------------
__global__ __launch_bounds__(256)
void cast_bf16_kernel(const float* __restrict__ s, unsigned short* __restrict__ d, int n) {
    int base = blockIdx.x * 1024 + threadIdx.x;
    #pragma unroll
    for (int k = 0; k < 4; ++k) {
        int i = base + k * 256;
        if (i < n) d[i] = f2bf(s[i]);
    }
}

// ---------------- WMMA bf16 GEMM: C[M,N] = act(A[M,K] @ W[N,K]^T + bias) ----------------
#define BM 128
#define BN 128
#define BK 64
#define LDK (BK + 8)   // 72 halves = 144B row stride = 32 + 4 DWORDs

#if USE_TDM
// Build a TDM D# for a 2D bf16 tile [BM x BK] at (row0, k0) of a [rows x Kdim]
// row-major tensor, landing in LDS with a 4-DWORD pad after every 32-DWORD row
// (reproducing the LDK=BK+8 padded layout).  CDNA5 ISA.
__device__ __forceinline__ void tdm_load_tile(const unsigned short* gbase,
                                              unsigned lds_addr,
                                              int row0, int k0,
                                              int Kdim, int rows) {
    unsigned long long ga = (unsigned long long)(uintptr_t)gbase
                          + ((unsigned long long)row0 * (unsigned)Kdim + (unsigned)k0) * 2ull;
    u32x4 g0;
    g0[0] = 1u;                                              // count=1 (valid user D#)
    g0[1] = lds_addr;                                        // LDS byte address
    g0[2] = (unsigned)ga;                                    // global_addr[31:0]
    g0[3] = (unsigned)((ga >> 32) & 0x01FFFFFFull)           // global_addr[56:32]
          | 0x80000000u;                                     // type=2 ("image")
    const unsigned td0 = (unsigned)Kdim;                     // tensor_dim0 (elements)
    const unsigned td1 = (unsigned)rows;                     // tensor_dim1 (rows)
    i32x8 g1;
    g1[0] = (int)((1u << 16)        // data_size = 2 bytes
                | (1u << 20)        // pad_enable
                | (4u << 22)        // pad_interval: 2^(4+1)=32 DWORDs
                | (3u << 25));      // pad_amount: 3+1 = 4 DWORDs
    g1[1] = (int)((td0 & 0xFFFFu) << 16);                    // tensor_dim0 low16 @ [63:48]
    g1[2] = (int)((td0 >> 16) | ((td1 & 0xFFFFu) << 16));    // dim0 hi16 | dim1 low16
    g1[3] = (int)((td1 >> 16) | ((unsigned)BK << 16));       // dim1 hi16 | tile_dim0=BK
    g1[4] = (int)((unsigned)BM);                             // tile_dim1=128, tile_dim2=0
    g1[5] = (int)td0;                                        // tensor_dim0_stride lo32 = Kdim
    g1[6] = 0;                                               // stride hi16 | dim1_stride lo16
    g1[7] = 0;
    i32x4 g2 = {1, 0, 0, 0};                                 // tensor_dim2=1, rest 0
    i32x4 g3 = {0, 0, 0, 0};
    i32x8 g4 = {0, 0, 0, 0, 0, 0, 0, 0};                     // extra group (6-arg form)
    __builtin_amdgcn_tensor_load_to_lds(g0, g1, g2, g3, g4, 0);
}
#endif

template<bool HAS_BIAS, bool ACT_TANH, bool OUT_BF16>
__global__ __launch_bounds__(256)
void gemm_bf16_wmma(const unsigned short* __restrict__ A,
                    const unsigned short* __restrict__ W,
                    const float* __restrict__ bias,
                    void* __restrict__ Cout,
                    int Mdim, int Ndim, int Kdim) {
    __shared__ unsigned short As[2][BM][LDK];
    __shared__ unsigned short Bs[2][BN][LDK];

    const int tid  = threadIdx.x;
    const int lane = tid & 31;
    const int wave = tid >> 5;        // 0..7
    const int wrow = wave >> 2;       // 0..1  -> 64-row slab
    const int wcol = wave & 3;        // 0..3  -> 32-col slab
    const int hi   = (lane >> 4) & 1;
    const int l15  = lane & 15;
    const int m0   = blockIdx.y * BM;
    const int n0   = blockIdx.x * BN;

    v8f acc[4][2];
    #pragma unroll
    for (int i = 0; i < 4; ++i)
        #pragma unroll
        for (int j = 0; j < 2; ++j) {
            v8f z = {0.f, 0.f, 0.f, 0.f, 0.f, 0.f, 0.f, 0.f};
            acc[i][j] = z;
        }

    const int ntiles = Kdim / BK;

#if USE_TDM
    if (wave == 0) {
        tdm_load_tile(A, (unsigned)(uintptr_t)&As[0][0][0], m0, 0, Kdim, Mdim);
        tdm_load_tile(W, (unsigned)(uintptr_t)&Bs[0][0][0], n0, 0, Kdim, Ndim);
    }
#else
    const int lr = tid >> 1;            // 0..127
    const int lc = (tid & 1) * 32;      // 0 or 32
    {
        const uint4* gA = (const uint4*)(A + (size_t)(m0 + lr) * Kdim + lc);
        uint4 a0 = gA[0], a1 = gA[1], a2 = gA[2], a3 = gA[3];
        const uint4* gB = (const uint4*)(W + (size_t)(n0 + lr) * Kdim + lc);
        uint4 b0 = gB[0], b1 = gB[1], b2 = gB[2], b3 = gB[3];
        uint4* sA = (uint4*)&As[0][lr][lc];
        sA[0] = a0; sA[1] = a1; sA[2] = a2; sA[3] = a3;
        uint4* sB = (uint4*)&Bs[0][lr][lc];
        sB[0] = b0; sB[1] = b1; sB[2] = b2; sB[3] = b3;
    }
#endif

    for (int t = 0; t < ntiles; ++t) {
        const int cur = t & 1;
#if USE_TDM
        if (wave == 0) __builtin_amdgcn_s_wait_tensorcnt((short)0);  // tile t landed
        __syncthreads();                                      // publish to all waves
        if (wave == 0 && (t + 1) < ntiles) {                  // overlap DMA of t+1 with math on t
            tdm_load_tile(A, (unsigned)(uintptr_t)&As[cur ^ 1][0][0],
                          m0, (t + 1) * BK, Kdim, Mdim);
            tdm_load_tile(W, (unsigned)(uintptr_t)&Bs[cur ^ 1][0][0],
                          n0, (t + 1) * BK, Kdim, Ndim);
        }
#else
        __syncthreads();
#endif

        #pragma unroll
        for (int ks = 0; ks < BK; ks += 32) {
            // B fragments: 32x16 (KxN) bf16, lane = column, K-halves [8h,8h+8) and [16+8h,24+8h)
            union { v16bf v; uint4 q[2]; } bf[2];
            #pragma unroll
            for (int j = 0; j < 2; ++j) {
                const int n = wcol * 32 + j * 16 + l15;
                const uint4* p = (const uint4*)&Bs[cur][n][ks + hi * 8];
                bf[j].q[0] = p[0];
                bf[j].q[1] = p[2];
            }
            #pragma unroll
            for (int i = 0; i < 4; ++i) {
                // A fragment: 16x32 (MxK) bf16 per ISA layout
                union { v16bf v; unsigned u[8]; } af;
                const int m = wrow * 64 + i * 16 + l15;
                #pragma unroll
                for (int v = 0; v < 8; ++v) {
                    const int kc = ks + ((v < 4) ? (hi * 8 + 2 * v)
                                                 : (16 + hi * 8 + 2 * (v - 4)));
                    af.u[v] = *(const unsigned*)&As[cur][m][kc];
                }
                #pragma unroll
                for (int j = 0; j < 2; ++j) {
                    acc[i][j] = __builtin_amdgcn_wmma_f32_16x16x32_bf16(
                        false, af.v, false, bf[j].v, (short)0, acc[i][j], false, false);
                }
            }
        }
        __syncthreads();   // all waves done reading before buffer reuse

#if !USE_TDM
        if (t + 1 < ntiles) {
            const int k0 = (t + 1) * BK;
            const int nxt = (t + 1) & 1;
            const uint4* gA = (const uint4*)(A + (size_t)(m0 + lr) * Kdim + k0 + lc);
            uint4 a0 = gA[0], a1 = gA[1], a2 = gA[2], a3 = gA[3];
            const uint4* gB = (const uint4*)(W + (size_t)(n0 + lr) * Kdim + k0 + lc);
            uint4 b0 = gB[0], b1 = gB[1], b2 = gB[2], b3 = gB[3];
            uint4* sA = (uint4*)&As[nxt][lr][lc];
            sA[0] = a0; sA[1] = a1; sA[2] = a2; sA[3] = a3;
            uint4* sB = (uint4*)&Bs[nxt][lr][lc];
            sB[0] = b0; sB[1] = b1; sB[2] = b2; sB[3] = b3;
        }
#endif
    }

    // epilogue: C layout -> VGPR v holds row (v + 8*hi), col = lane&15
    #pragma unroll
    for (int i = 0; i < 4; ++i) {
        #pragma unroll
        for (int j = 0; j < 2; ++j) {
            const int col = n0 + wcol * 32 + j * 16 + l15;
            const float bv = HAS_BIAS ? bias[col] : 0.0f;
            #pragma unroll
            for (int v = 0; v < 8; ++v) {
                const int row = m0 + wrow * 64 + i * 16 + v + hi * 8;
                float xv = acc[i][j][v] + bv;
                if (ACT_TANH) xv = tanhf(xv);
                if (OUT_BF16)
                    ((unsigned short*)Cout)[(size_t)row * Ndim + col] = f2bf(xv);
                else
                    ((float*)Cout)[(size_t)row * Ndim + col] = xv;
            }
        }
    }
}

// ---------------- LIF scan: sequential over T, one thread per (b,d) ----------------
__global__ __launch_bounds__(256)
void lif_scan_kernel(float* __restrict__ ic_spike,    // in: input_current, out: spike_out
                     const float* __restrict__ xt,
                     const float* __restrict__ noise,  // [T,B,D]
                     const float* __restrict__ res_scale) {
    const int gid = blockIdx.x * 256 + threadIdx.x;   // 0 .. B*D-1
    const int b = gid >> 10;
    const int d = gid & 1023;
    const float r = 1.0f / (1.0f + expf(-res_scale[0]));
    const float omr = 1.0f - r;
    float mem = 0.0f;
    for (int t = 0; t < T_; ++t) {
        const size_t idx = ((size_t)b * T_ + t) * D_ + d;
        const float cur = ic_spike[idx];
        const float nz  = noise[((size_t)t * B_ + b) * D_ + d];
        mem = 0.95f * mem + cur + nz * 0.00075f + 1e-6f;   // 0.005*THRESHOLD
        const float spk = (mem >= 0.15f) ? 1.0f : 0.0f;
        mem = mem * (1.0f - spk) - 0.05f * spk;
        const float raw = spk * cur * 0.5f;
        ic_spike[idx] = tanhf((r * raw + omr * xt[idx]) * 0.5f);
    }
}

// ---------------- layernorm (row of 1024), bf16 out ----------------
__global__ __launch_bounds__(256)
void ln_bf16_kernel(const float* __restrict__ in, const float* __restrict__ g,
                    const float* __restrict__ b, unsigned short* __restrict__ out) {
    __shared__ float rs[256], rq[256];
    const int tid = threadIdx.x;
    const size_t base = (size_t)blockIdx.x * D_;
    float v[4], s = 0.f, q = 0.f;
    #pragma unroll
    for (int k = 0; k < 4; ++k) {
        v[k] = in[base + tid + k * 256];
        s += v[k]; q += v[k] * v[k];
    }
    rs[tid] = s; rq[tid] = q; __syncthreads();
    for (int st = 128; st > 0; st >>= 1) {
        if (tid < st) { rs[tid] += rs[tid + st]; rq[tid] += rq[tid + st]; }
        __syncthreads();
    }
    const float mean = rs[0] * (1.0f / D_);
    const float var  = rq[0] * (1.0f / D_) - mean * mean;
    const float rstd = rsqrtf(var + 1e-5f);
    #pragma unroll
    for (int k = 0; k < 4; ++k) {
        const int c = tid + k * 256;
        out[base + c] = f2bf((v[k] - mean) * rstd * g[c] + b[c]);
    }
}

// ---------------- fused block_ln -> nan_to_num -> pre_ln, bf16 out ----------------
__global__ __launch_bounds__(256)
void double_ln_kernel(const float* __restrict__ in,
                      const float* __restrict__ g1, const float* __restrict__ b1,
                      const float* __restrict__ g2, const float* __restrict__ b2,
                      unsigned short* __restrict__ out) {
    __shared__ float rs[256], rq[256];
    const int tid = threadIdx.x;
    const size_t base = (size_t)blockIdx.x * D_;
    float v[4], s = 0.f, q = 0.f;
    #pragma unroll
    for (int k = 0; k < 4; ++k) {
        v[k] = in[base + tid + k * 256];
        s += v[k]; q += v[k] * v[k];
    }
    rs[tid] = s; rq[tid] = q; __syncthreads();
    for (int st = 128; st > 0; st >>= 1) {
        if (tid < st) { rs[tid] += rs[tid + st]; rq[tid] += rq[tid + st]; }
        __syncthreads();
    }
    float mean = rs[0] * (1.0f / D_);
    float rstd = rsqrtf(rq[0] * (1.0f / D_) - mean * mean + 1e-5f);
    s = 0.f; q = 0.f;
    #pragma unroll
    for (int k = 0; k < 4; ++k) {
        const int c = tid + k * 256;
        v[k] = nan2num((v[k] - mean) * rstd * g1[c] + b1[c]);
        s += v[k]; q += v[k] * v[k];
    }
    __syncthreads();
    rs[tid] = s; rq[tid] = q; __syncthreads();
    for (int st = 128; st > 0; st >>= 1) {
        if (tid < st) { rs[tid] += rs[tid + st]; rq[tid] += rq[tid + st]; }
        __syncthreads();
    }
    mean = rs[0] * (1.0f / D_);
    rstd = rsqrtf(rq[0] * (1.0f / D_) - mean * mean + 1e-5f);
    #pragma unroll
    for (int k = 0; k < 4; ++k) {
        const int c = tid + k * 256;
        out[base + c] = f2bf((v[k] - mean) * rstd * g2[c] + b2[c]);
    }
}

// ---------------- final: nan_to_num(attn) + noise*std -> block_ln -> f32 out ----------------
__global__ __launch_bounds__(256)
void final_ln_kernel(const float* __restrict__ in, const float* __restrict__ nq,
                     const float* __restrict__ qscale,
                     const float* __restrict__ g, const float* __restrict__ b,
                     float* __restrict__ out) {
    __shared__ float rs[256], rq[256];
    const int tid = threadIdx.x;
    const size_t base = (size_t)blockIdx.x * D_;
    const float nstd = fminf(fmaxf(fabsf(qscale[0]), 1e-6f), 0.003f);
    float v[4], s = 0.f, q = 0.f;
    #pragma unroll
    for (int k = 0; k < 4; ++k) {
        const size_t i = base + tid + k * 256;
        v[k] = nan2num(in[i]) + nq[i] * nstd;
        s += v[k]; q += v[k] * v[k];
    }
    rs[tid] = s; rq[tid] = q; __syncthreads();
    for (int st = 128; st > 0; st >>= 1) {
        if (tid < st) { rs[tid] += rs[tid + st]; rq[tid] += rq[tid + st]; }
        __syncthreads();
    }
    const float mean = rs[0] * (1.0f / D_);
    const float rstd = rsqrtf(rq[0] * (1.0f / D_) - mean * mean + 1e-5f);
    #pragma unroll
    for (int k = 0; k < 4; ++k) {
        const int c = tid + k * 256;
        out[base + c] = (v[k] - mean) * rstd * g[c] + b[c];
    }
}

// ---------------- entangled projection: [M,D] x ent_left[H,hd,R] -> [M,H*R], clip ----------------
__global__ __launch_bounds__(256)
void ent_proj_kernel(const float* __restrict__ P, const float* __restrict__ el,
                     float* __restrict__ out) {
    const int gid = blockIdx.x * 256 + threadIdx.x;   // M*64
    const int m = gid >> 6, hr = gid & 63, h = hr >> 4, r = hr & 15;
    const float* p = P + (size_t)m * D_ + h * HD_;
    const float* w = el + ((size_t)h * HD_) * R_ + r;
    float acc = 0.f;
    for (int d = 0; d < HD_; ++d) acc += p[d] * w[(size_t)d * R_];
    out[gid] = fminf(fmaxf(acc, -10.0f), 10.0f);
}

// ---------------- scores over T + softmax: per (b,h) block ----------------
__global__ __launch_bounds__(256)
void scores_softmax_kernel(const float* __restrict__ q, const float* __restrict__ k,
                           float* __restrict__ probs) {
    const int bh = blockIdx.x;                 // 0..63
    const int b = bh >> 2, h = bh & 3;
    const int tid = threadIdx.x;
    const int r = tid >> 4, s = tid & 15;
    const float* qp = q + (size_t)b * T_ * 64 + h * 16 + r;
    const float* kp = k + (size_t)b * T_ * 64 + h * 16 + s;
    float acc = 0.f;
    for (int t = 0; t < T_; ++t) acc += qp[(size_t)t * 64] * kp[(size_t)t * 64];
    __shared__ float sc[16][16];
    sc[r][s] = acc * 0.25f;                    // 1/sqrt(R), R=16
    __syncthreads();
    if (tid < 16) {
        float mx = -3.4e38f;
        #pragma unroll
        for (int j = 0; j < 16; ++j) mx = fmaxf(mx, sc[tid][j]);
        float e[16], sum = 0.f;
        #pragma unroll
        for (int j = 0; j < 16; ++j) { e[j] = expf(sc[tid][j] - mx); sum += e[j]; }
        const float inv = 1.0f / sum;
        #pragma unroll
        for (int j = 0; j < 16; ++j)
            probs[((size_t)bh * 16 + tid) * 16 + j] = e[j] * inv;
    }
}

// ---------------- attn apply: probs x v -> ao, ao x ent_right -> bf16 [M,D] ----------------
__global__ __launch_bounds__(256)
void attn_apply_kernel(const float* __restrict__ probs, const float* __restrict__ v,
                       const float* __restrict__ er, unsigned short* __restrict__ out) {
    const int m = blockIdx.x;
    const int bb = m >> 9;                     // m / T_
    const int tid = threadIdx.x;
    __shared__ float ao[64];
    if (tid < 64) {
        const int h = tid >> 4, r = tid & 15;
        const float* pr = probs + ((size_t)(bb * 4 + h) * 16 + r) * 16;
        const float* vv = v + (size_t)m * 64 + h * 16;
        float acc = 0.f;
        #pragma unroll
        for (int s = 0; s < 16; ++s) acc += pr[s] * vv[s];
        ao[tid] = acc;
    }
    __syncthreads();
    #pragma unroll
    for (int j = 0; j < 4; ++j) {
        const int c = tid + j * 256;
        const int h = c >> 8, dh = c & 255;
        float acc = 0.f;
        #pragma unroll
        for (int r = 0; r < 16; ++r)
            acc += ao[h * 16 + r] * er[((size_t)(h * 16 + r)) * HD_ + dh];
        out[(size_t)m * D_ + c] = f2bf(acc);
    }
}

// ---------------- fractal compression: out[b,d] = sum_s x[b,s,d]*basis[s,d] ----------------
__global__ __launch_bounds__(256)
void compress_kernel(const float* __restrict__ x, const float* __restrict__ basis,
                     float* __restrict__ out) {
    const int gid = blockIdx.x * 256 + threadIdx.x;   // B*D
    const int b = gid >> 10, d = gid & 1023;
    float acc = 0.f;
    for (int s = 0; s < T_; ++s)
        acc += x[((size_t)b * T_ + s) * D_ + d] * basis[(size_t)s * D_ + d];
    out[gid] = acc;
}

// ==================================================================================
extern "C" void kernel_launch(void* const* d_in, const int* in_sizes, int n_in,
                              void* d_out, int out_size, void* d_ws, size_t ws_size,
                              hipStream_t stream) {
    (void)in_sizes; (void)n_in; (void)out_size; (void)ws_size;
    const float* x         = (const float*)d_in[0];
    const float* noise_lif = (const float*)d_in[1];
    const float* noise_q   = (const float*)d_in[2];
    const float* lif_w     = (const float*)d_in[3];
    const float* res_scale = (const float*)d_in[4];
    const float* hebb_g    = (const float*)d_in[5];
    const float* hebb_b    = (const float*)d_in[6];
    const float* hebb_w    = (const float*)d_in[7];
    const float* hproj_w   = (const float*)d_in[8];
    const float* hproj_b   = (const float*)d_in[9];
    const float* blk_g     = (const float*)d_in[10];
    const float* blk_b     = (const float*)d_in[11];
    const float* pre_g     = (const float*)d_in[12];
    const float* pre_b     = (const float*)d_in[13];
    const float* q_w       = (const float*)d_in[14];
    const float* q_b       = (const float*)d_in[15];
    const float* k_w       = (const float*)d_in[16];
    const float* k_b       = (const float*)d_in[17];
    const float* v_w       = (const float*)d_in[18];
    const float* v_b       = (const float*)d_in[19];
    const float* ent_left  = (const float*)d_in[20];
    const float* ent_right = (const float*)d_in[21];
    const float* aproj_w   = (const float*)d_in[22];
    const float* aproj_b   = (const float*)d_in[23];
    const float* qscale    = (const float*)d_in[24];
    const float* basis     = (const float*)d_in[25];

    // workspace layout
    char* w = (char*)d_ws;
    const size_t SZ_MD_F = (size_t)M_ * D_ * 4;     // 32 MB
    const size_t SZ_MD_B = (size_t)M_ * D_ * 2;     // 16 MB
    float*          buf1f = (float*)(w);                               // xt_f32 / hebb2 / attn_pre
    unsigned short* buf2b = (unsigned short*)(w + SZ_MD_F);            // xt_bf16 / h / xn
    float*          buf3f = (float*)(w + SZ_MD_F + SZ_MD_B);           // ic->spike / P / ao2(bf16)
    unsigned short* buf4b = (unsigned short*)(w + 2*SZ_MD_F + SZ_MD_B);// hebb1 bf16 [M, D/2]
    char* p5 = w + 2*SZ_MD_F + SZ_MD_B + (size_t)M_ * (D_/2) * 2;
    float* qs    = (float*)p5;
    float* ks    = qs + (size_t)M_ * 64;
    float* vs    = ks + (size_t)M_ * 64;
    float* probs = vs + (size_t)M_ * 64;
    unsigned short* wslot = (unsigned short*)(probs + B_ * H_ * R_ * R_); // 2 MB rotating
    unsigned short* ao2b  = (unsigned short*)buf3f;

    const dim3 blk(256);
    const dim3 gNN(D_ / BN, M_ / BM);        // N=1024
    const dim3 gNH(512 / BN, M_ / BM);       // N=512

    // 1) xt = tanh(0.5*x)
    prep_xt_kernel<<<M_ * D_ / 1024, blk, 0, stream>>>(x, buf1f, buf2b);

    // 2) input_current = tanh(xt @ lif_w^T)
    cast_bf16_kernel<<<(D_ * D_) / 1024, blk, 0, stream>>>(lif_w, wslot, D_ * D_);
    gemm_bf16_wmma<false, true, false><<<gNN, blk, 0, stream>>>(buf2b, wslot, nullptr, buf3f, M_, D_, D_);

    // 3) LIF scan -> spike_out (in place in buf3f)
    lif_scan_kernel<<<B_ * D_ / 256, blk, 0, stream>>>(buf3f, buf1f, noise_lif, res_scale);

    // 4) h = LN(spike_out) -> bf16
    ln_bf16_kernel<<<M_, blk, 0, stream>>>(buf3f, hebb_g, hebb_b, buf2b);

    // 5) hebb1 = h @ hebb_w^T  [M,512] bf16
    cast_bf16_kernel<<<(512 * D_) / 1024, blk, 0, stream>>>(hebb_w, wslot, 512 * D_);
    gemm_bf16_wmma<false, false, true><<<gNH, blk, 0, stream>>>(buf2b, wslot, nullptr, buf4b, M_, 512, D_);

    // 6) hebb2 = hebb1 @ hproj_w^T + b  [M,1024] f32
    cast_bf16_kernel<<<(D_ * 512) / 1024, blk, 0, stream>>>(hproj_w, wslot, D_ * 512);
    gemm_bf16_wmma<true, false, false><<<gNN, blk, 0, stream>>>(buf4b, wslot, hproj_b, buf1f, M_, D_, 512);

    // 7) xn = pre_ln(nan_to_num(block_ln(hebb2))) -> bf16
    double_ln_kernel<<<M_, blk, 0, stream>>>(buf1f, blk_g, blk_b, pre_g, pre_b, buf2b);

    // 8) q/k/v projections + ent_left contraction (+clip)
    cast_bf16_kernel<<<(D_ * D_) / 1024, blk, 0, stream>>>(q_w, wslot, D_ * D_);
    gemm_bf16_wmma<true, false, false><<<gNN, blk, 0, stream>>>(buf2b, wslot, q_b, buf3f, M_, D_, D_);
    ent_proj_kernel<<<M_ * 64 / 256, blk, 0, stream>>>(buf3f, ent_left, qs);

    cast_bf16_kernel<<<(D_ * D_) / 1024, blk, 0, stream>>>(k_w, wslot, D_ * D_);
    gemm_bf16_wmma<true, false, false><<<gNN, blk, 0, stream>>>(buf2b, wslot, k_b, buf3f, M_, D_, D_);
    ent_proj_kernel<<<M_ * 64 / 256, blk, 0, stream>>>(buf3f, ent_left, ks);

    cast_bf16_kernel<<<(D_ * D_) / 1024, blk, 0, stream>>>(v_w, wslot, D_ * D_);
    gemm_bf16_wmma<true, false, false><<<gNN, blk, 0, stream>>>(buf2b, wslot, v_b, buf3f, M_, D_, D_);
    ent_proj_kernel<<<M_ * 64 / 256, blk, 0, stream>>>(buf3f, ent_left, vs);

    // 9) scores + softmax, then probs*v, *ent_right -> ao2 bf16
    scores_softmax_kernel<<<B_ * H_, blk, 0, stream>>>(qs, ks, probs);
    attn_apply_kernel<<<M_, blk, 0, stream>>>(probs, vs, ent_right, ao2b);

    // 10) attn_out = ao2 @ aproj_w^T + b
    cast_bf16_kernel<<<(D_ * D_) / 1024, blk, 0, stream>>>(aproj_w, wslot, D_ * D_);
    gemm_bf16_wmma<true, false, false><<<gNN, blk, 0, stream>>>(ao2b, wslot, aproj_b, buf1f, M_, D_, D_);

    // 11) gated = LN(nan_to_num(attn_out) + noise_q*std) -> d_out[0 : M*D]
    final_ln_kernel<<<M_, blk, 0, stream>>>(buf1f, noise_q, qscale, blk_g, blk_b, (float*)d_out);

    // 12) compressed -> d_out[M*D : M*D + B*D]
    compress_kernel<<<B_ * D_ / 256, blk, 0, stream>>>(x, basis, (float*)d_out + (size_t)M_ * D_);
}
